// CustomMultiheadAttention_45689862094998
// MI455X (gfx1250) — matrix operand order repaired and checked by reading, not compiled, not run on hardware
//
#include <hip/hip_runtime.h>
#include <hip/hip_bf16.h>

#define BATCH  2
#define S_LEN  2048
#define EMB    1024
#define HEADS  16
#define HD     64
#define MROWS  (BATCH * S_LEN)   // 4096

typedef __attribute__((ext_vector_type(16))) __bf16 bf16x16;
typedef __attribute__((ext_vector_type(8)))  __bf16 bf16x8;
typedef __attribute__((ext_vector_type(4)))  __bf16 bf16x4;
typedef __attribute__((ext_vector_type(8)))  float  f32x8;

__device__ __forceinline__ f32x8 zero8() {
    f32x8 z;
#pragma unroll
    for (int i = 0; i < 8; ++i) z[i] = 0.0f;
    return z;
}

__device__ __forceinline__ f32x8 wmma_bf16(bf16x16 a, bf16x16 b, f32x8 c) {
    // v_wmma_f32_16x16x32_bf16: D = A(16x32) * B(32x16) + C(16x16 f32)
    return __builtin_amdgcn_wmma_f32_16x16x32_bf16(
        /*neg_a=*/false, a, /*neg_b=*/false, b,
        /*c_mod=*/(short)0, c, /*reuse_a=*/false, /*reuse_b=*/false);
}

// A-matrix fragment (16x32 bf16, row-major source):
// lanes 0-15 (m = lane): K = k0+0..7 (elems 0-7), k0+16..23 (elems 8-15)
// lanes 16-31 (m = lane-16): K = k0+8..15, k0+24..31
__device__ __forceinline__ bf16x16 load_frag_a(const __bf16* __restrict__ base,
                                               int ld, int row0, int k0, int lane) {
    const int m = lane & 15, sel = lane >> 4;
    const __bf16* p = base + (size_t)(row0 + m) * ld + k0 + 8 * sel;
    bf16x8 lo = *(const bf16x8*)(p);
    bf16x8 hi = *(const bf16x8*)(p + 16);
    bf16x16 r;
#pragma unroll
    for (int i = 0; i < 8; ++i) { r[i] = lo[i]; r[i + 8] = hi[i]; }
    return r;
}

// B-matrix fragment (32x16 bf16): lane n (n = lane&15) holds column n,
// K = k0 + 8*(lane>>4) + 0..7 (elems 0-7), k0 + 16 + 8*(lane>>4) + 0..7 (elems 8-15).
__device__ __forceinline__ bf16x16 load_frag_b(const __bf16* __restrict__ base,
                                               int ld, int n0, int k0, int lane) {
    const int n = lane & 15, sel = lane >> 4;
    const __bf16* p = base + (size_t)(n0 + n) * ld + k0 + 8 * sel;
    bf16x8 lo = *(const bf16x8*)(p);
    bf16x8 hi = *(const bf16x8*)(p + 16);
    bf16x16 r;
#pragma unroll
    for (int i = 0; i < 8; ++i) { r[i] = lo[i]; r[i + 8] = hi[i]; }
    return r;
}

// Async global -> LDS copy of one 16B chunk per lane (CDNA5 ASYNCcnt path).
__device__ __forceinline__ void async_copy_b128(const __bf16* gsrc, __bf16* lds_dst) {
    unsigned ldso = (unsigned)(uintptr_t)lds_dst;          // LDS byte offset (low 32b)
    unsigned long long ga = (unsigned long long)(uintptr_t)gsrc;
    asm volatile("global_load_async_to_lds_b128 %0, %1, off"
                 :: "v"(ldso), "v"(ga) : "memory");
}
__device__ __forceinline__ void wait_async_zero() {
    asm volatile("s_wait_asynccnt 0" ::: "memory");
}

// ---------------------------------------------------------------- convert
__global__ void cvt_f32_to_bf16_kernel(const float4* __restrict__ src,
                                       bf16x4* __restrict__ dst, int n4) {
    int i = blockIdx.x * blockDim.x + threadIdx.x;
    if (i < n4) {
        float4 f = src[i];
        bf16x4 o;
        o[0] = (__bf16)f.x; o[1] = (__bf16)f.y;
        o[2] = (__bf16)f.z; o[3] = (__bf16)f.w;
        dst[i] = o;
    }
}

// ---------------------------------------------------------------- GEMM: out = X @ W^T + bias
// mode 0: store bf16 as [B,H,S,D]; mode 1: store bf16 as [B,H,D,S] (V transposed);
// mode 2: store fp32 row-major [M, EMB].
__global__ __launch_bounds__(128) void gemm_bias_kernel(
    const __bf16* __restrict__ X, const __bf16* __restrict__ W,
    const float* __restrict__ bias,
    __bf16* __restrict__ out_bf, float* __restrict__ out_f32,
    float scale, int mode) {
    const int lane = threadIdx.x & 31;
    const int wave = threadIdx.x >> 5;
    const int m0 = blockIdx.x * 128 + (wave & 1) * 64;
    const int n0 = blockIdx.y * 128 + (wave >> 1) * 64;

    f32x8 acc[4][4];
#pragma unroll
    for (int i = 0; i < 4; ++i)
#pragma unroll
        for (int j = 0; j < 4; ++j) acc[i][j] = zero8();

    for (int k0 = 0; k0 < EMB; k0 += 32) {
        bf16x16 a[4], b[4];
#pragma unroll
        for (int i = 0; i < 4; ++i) a[i] = load_frag_a(X, EMB, m0 + 16 * i, k0, lane);
#pragma unroll
        for (int j = 0; j < 4; ++j) b[j] = load_frag_b(W, EMB, n0 + 16 * j, k0, lane);
#pragma unroll
        for (int i = 0; i < 4; ++i)
#pragma unroll
            for (int j = 0; j < 4; ++j) acc[i][j] = wmma_bf16(a[i], b[j], acc[i][j]);
    }

    const int col = lane & 15, sel = lane >> 4;
#pragma unroll
    for (int j = 0; j < 4; ++j) {
        const float bj = bias[n0 + 16 * j + col];
#pragma unroll
        for (int i = 0; i < 4; ++i) {
#pragma unroll
            for (int r = 0; r < 8; ++r) {
                float v = (acc[i][j][r] + bj) * scale;
                int m = m0 + 16 * i + r + 8 * sel;
                int n = n0 + 16 * j + col;
                if (mode == 2) {
                    out_f32[(size_t)m * EMB + n] = v;
                } else {
                    int bb = m >> 11, s = m & (S_LEN - 1);
                    int h = n >> 6, d = n & (HD - 1);
                    size_t bh = (size_t)bb * HEADS + h;
                    if (mode == 0)
                        out_bf[(bh * S_LEN + s) * HD + d] = (__bf16)v;
                    else
                        out_bf[(bh * HD + d) * S_LEN + s] = (__bf16)v;
                }
            }
        }
    }
}

// ---------------------------------------------------------------- softmax stats (online)
// Transposed orientation: S^T[k][q] = sum_d K[k,d]*Q[q,d]; A=K rows, B=Q rows.
// C layout: reg r + 8*sel = k index, lane&15 = q column -> row reduction is per-lane.
__global__ __launch_bounds__(256) void softmax_stats_kernel(
    const __bf16* __restrict__ Q, const __bf16* __restrict__ K,
    float* __restrict__ Mrow, float* __restrict__ Linv) {
    const int lane = threadIdx.x & 31;
    const int wave = threadIdx.x >> 5;
    const int bh = blockIdx.x;
    const int q0 = blockIdx.y * 512 + wave * 64;
    const __bf16* Qb = Q + (size_t)bh * S_LEN * HD;
    const __bf16* Kb = K + (size_t)bh * S_LEN * HD;

    bf16x16 qb[4][2];
#pragma unroll
    for (int t = 0; t < 4; ++t)
#pragma unroll
        for (int d = 0; d < 2; ++d)
            qb[t][d] = load_frag_b(Qb, HD, q0 + 16 * t, 32 * d, lane);

    float mrun[4], lrun[4];
#pragma unroll
    for (int t = 0; t < 4; ++t) { mrun[t] = -3.0e38f; lrun[t] = 0.0f; }

    for (int kb = 0; kb < S_LEN; kb += 16) {
        bf16x16 a0 = load_frag_a(Kb, HD, kb, 0, lane);
        bf16x16 a1 = load_frag_a(Kb, HD, kb, 32, lane);
#pragma unroll
        for (int t = 0; t < 4; ++t) {
            f32x8 s = wmma_bf16(a0, qb[t][0], zero8());
            s = wmma_bf16(a1, qb[t][1], s);
            float mt = s[0];
#pragma unroll
            for (int r = 1; r < 8; ++r) mt = fmaxf(mt, s[r]);
            float mnew = fmaxf(mrun[t], mt);
            float sum = 0.0f;
#pragma unroll
            for (int r = 0; r < 8; ++r) sum += __expf(s[r] - mnew);
            lrun[t] = lrun[t] * __expf(mrun[t] - mnew) + sum;
            mrun[t] = mnew;
        }
    }
    // combine the two k-halves held by lane pairs (lane, lane^16)
#pragma unroll
    for (int t = 0; t < 4; ++t) {
        float mo = __shfl_xor(mrun[t], 16, 32);
        float lo = __shfl_xor(lrun[t], 16, 32);
        float m = fmaxf(mrun[t], mo);
        float l = lrun[t] * __expf(mrun[t] - m) + lo * __expf(mo - m);
        if (lane < 16) {
            Mrow[(size_t)bh * S_LEN + q0 + 16 * t + lane] = m;
            Linv[(size_t)bh * S_LEN + q0 + 16 * t + lane] = 1.0f / l;
        }
    }
}

// ---------------------------------------------------------------- attn mean over heads
// Normal orientation: row = q (in regs), col = k (in lanes) -> coalesced fp32 stores.
__global__ __launch_bounds__(128) void attn_mean_kernel(
    const __bf16* __restrict__ Q, const __bf16* __restrict__ K,
    const float* __restrict__ Mrow, const float* __restrict__ Linv,
    float* __restrict__ out2) {
    const int lane = threadIdx.x & 31;
    const int wave = threadIdx.x >> 5;
    const int b = blockIdx.z;
    const int q0 = blockIdx.x * 128 + (wave & 1) * 64;
    const int k0 = blockIdx.y * 128 + (wave >> 1) * 64;
    const int col = lane & 15, sel = lane >> 4;

    f32x8 macc[4][4];
#pragma unroll
    for (int i = 0; i < 4; ++i)
#pragma unroll
        for (int j = 0; j < 4; ++j) macc[i][j] = zero8();

    for (int h = 0; h < HEADS; ++h) {
        const size_t bh = (size_t)b * HEADS + h;
        const __bf16* Qb = Q + bh * S_LEN * HD;
        const __bf16* Kb = K + bh * S_LEN * HD;
        bf16x16 qa[4][2], kf[4][2];
#pragma unroll
        for (int t = 0; t < 4; ++t)
#pragma unroll
            for (int d = 0; d < 2; ++d) {
                qa[t][d] = load_frag_a(Qb, HD, q0 + 16 * t, 32 * d, lane);
                kf[t][d] = load_frag_b(Kb, HD, k0 + 16 * t, 32 * d, lane);
            }
#pragma unroll
        for (int i = 0; i < 4; ++i) {
            float mr[8], li[8];
#pragma unroll
            for (int r = 0; r < 8; ++r) {
                int qq = q0 + 16 * i + 8 * sel + r;
                mr[r] = Mrow[bh * S_LEN + qq];
                li[r] = Linv[bh * S_LEN + qq];
            }
#pragma unroll
            for (int j = 0; j < 4; ++j) {
                f32x8 s = wmma_bf16(qa[i][0], kf[j][0], zero8());
                s = wmma_bf16(qa[i][1], kf[j][1], s);
#pragma unroll
                for (int r = 0; r < 8; ++r)
                    macc[i][j][r] += __expf(s[r] - mr[r]) * li[r];
            }
        }
    }
    const float inv_h = 1.0f / (float)HEADS;
#pragma unroll
    for (int i = 0; i < 4; ++i)
#pragma unroll
        for (int j = 0; j < 4; ++j)
#pragma unroll
            for (int r = 0; r < 8; ++r) {
                int qq = q0 + 16 * i + 8 * sel + r;
                int kk = k0 + 16 * j + col;
                out2[(size_t)b * S_LEN * S_LEN + (size_t)qq * S_LEN + kk] =
                    macc[i][j][r] * inv_h;
            }
}

// ---------------------------------------------------------------- out = softmax(QK^T) @ V
// K (32x64) and V (64x32) tiles are identical for all 4 waves -> stage them in
// LDS once per block via async global->LDS (ASYNCcnt), then all waves read
// fragments from LDS. P tiles bounce through wave-private LDS (C->A relayout).
__global__ __launch_bounds__(128) void attn_v_kernel(
    const __bf16* __restrict__ Q, const __bf16* __restrict__ K,
    const __bf16* __restrict__ Vt,
    const float* __restrict__ Mrow, const float* __restrict__ Linv,
    __bf16* __restrict__ outh) {
    __shared__ __align__(64) __bf16 lds_p[4][64 * 32];
    __shared__ __align__(64) __bf16 lds_k[32 * 64];   // [k row][d]
    __shared__ __align__(64) __bf16 lds_v[64 * 32];   // [d row][k]
    const int tid = threadIdx.x;
    const int lane = tid & 31;
    const int wave = tid >> 5;
    const int bh = blockIdx.x;
    const int b = bh >> 4, h = bh & 15;
    const int q0 = blockIdx.y * 256 + wave * 64;
    const int col = lane & 15, sel = lane >> 4;
    const __bf16* Qb = Q + (size_t)bh * S_LEN * HD;
    const __bf16* Kb = K + (size_t)bh * S_LEN * HD;
    const __bf16* Vb = Vt + (size_t)bh * HD * S_LEN;
    __bf16* myp = &lds_p[wave][0];

    bf16x16 qa[4][2];
#pragma unroll
    for (int i = 0; i < 4; ++i)
#pragma unroll
        for (int d = 0; d < 2; ++d)
            qa[i][d] = load_frag_a(Qb, HD, q0 + 16 * i, 32 * d, lane);

    float mr[4][8], li[4][8];
#pragma unroll
    for (int i = 0; i < 4; ++i)
#pragma unroll
        for (int r = 0; r < 8; ++r) {
            int qq = q0 + 16 * i + 8 * sel + r;
            mr[i][r] = Mrow[(size_t)bh * S_LEN + qq];
            li[i][r] = Linv[(size_t)bh * S_LEN + qq];
        }

    f32x8 oacc[4][4];
#pragma unroll
    for (int i = 0; i < 4; ++i)
#pragma unroll
        for (int j = 0; j < 4; ++j) oacc[i][j] = zero8();

    for (int kb = 0; kb < S_LEN; kb += 32) {
        // --- stage K tile (rows kb..kb+31 x 64d) and V tile (64d x k kb..kb+31)
        // 256 + 256 16B chunks, 128 threads -> 2+2 async issues per thread.
#pragma unroll
        for (int it = 0; it < 2; ++it) {
            int c = tid + 128 * it;
            async_copy_b128(Kb + (size_t)(kb + (c >> 3)) * HD + (c & 7) * 8,
                            lds_k + c * 8);
            async_copy_b128(Vb + (size_t)(c >> 2) * S_LEN + kb + (c & 3) * 8,
                            lds_v + c * 8);
        }
        wait_async_zero();
        __syncthreads();

        // --- S = Q K^T, P = exp(S - m) / l -> wave-private LDS (bf16, A-layout src)
#pragma unroll
        for (int ks = 0; ks < 2; ++ks) {
            bf16x16 k0f = load_frag_b(lds_k, HD, 16 * ks, 0, lane);
            bf16x16 k1f = load_frag_b(lds_k, HD, 16 * ks, 32, lane);
#pragma unroll
            for (int i = 0; i < 4; ++i) {
                f32x8 s = wmma_bf16(qa[i][0], k0f, zero8());
                s = wmma_bf16(qa[i][1], k1f, s);
#pragma unroll
                for (int r = 0; r < 8; ++r) {
                    float p = __expf(s[r] - mr[i][r]) * li[i][r];
                    myp[(16 * i + r + 8 * sel) * 32 + 16 * ks + col] = (__bf16)p;
                }
            }
        }
        // --- oacc += P @ V
        bf16x16 vb[4];
#pragma unroll
        for (int j = 0; j < 4; ++j) vb[j] = load_frag_b(lds_v, 32, 16 * j, 0, lane);
#pragma unroll
        for (int i = 0; i < 4; ++i) {
            bf16x16 pa = load_frag_a(myp, 32, 16 * i, 0, lane);
#pragma unroll
            for (int j = 0; j < 4; ++j) oacc[i][j] = wmma_bf16(pa, vb[j], oacc[i][j]);
        }
        __syncthreads();   // protect lds_k/lds_v before next async overwrite
    }
#pragma unroll
    for (int i = 0; i < 4; ++i)
#pragma unroll
        for (int j = 0; j < 4; ++j)
#pragma unroll
            for (int r = 0; r < 8; ++r) {
                int sidx = q0 + 16 * i + r + 8 * sel;
                int d = h * HD + 16 * j + col;
                outh[((size_t)b * S_LEN + sidx) * EMB + d] = (__bf16)oacc[i][j][r];
            }
}

// ---------------------------------------------------------------- launch
extern "C" void kernel_launch(void* const* d_in, const int* in_sizes, int n_in,
                              void* d_out, int out_size, void* d_ws, size_t ws_size,
                              hipStream_t stream) {
    const float* query = (const float*)d_in[0];
    const float* Wq = (const float*)d_in[1];
    const float* bq = (const float*)d_in[2];
    const float* Wk = (const float*)d_in[3];
    const float* bk = (const float*)d_in[4];
    const float* Wv = (const float*)d_in[5];
    const float* bv = (const float*)d_in[6];
    const float* Wo = (const float*)d_in[7];
    const float* bo = (const float*)d_in[8];

    char* ws = (char*)d_ws;
    size_t off = 0;
    auto walloc = [&](size_t bytes) {
        char* p = ws + off;
        off = (off + bytes + 255) & ~(size_t)255;
        return p;
    };
    __bf16* qbf  = (__bf16*)walloc((size_t)MROWS * EMB * 2);
    __bf16* wqb  = (__bf16*)walloc((size_t)EMB * EMB * 2);
    __bf16* wkb  = (__bf16*)walloc((size_t)EMB * EMB * 2);
    __bf16* wvb  = (__bf16*)walloc((size_t)EMB * EMB * 2);
    __bf16* wob  = (__bf16*)walloc((size_t)EMB * EMB * 2);
    __bf16* Qh   = (__bf16*)walloc((size_t)MROWS * EMB * 2);   // [B,H,S,D] scaled
    __bf16* Kh   = (__bf16*)walloc((size_t)MROWS * EMB * 2);   // [B,H,S,D]
    __bf16* Vth  = (__bf16*)walloc((size_t)MROWS * EMB * 2);   // [B,H,D,S]
    float*  Mst  = (float*)walloc((size_t)BATCH * HEADS * S_LEN * 4);
    float*  Lst  = (float*)walloc((size_t)BATCH * HEADS * S_LEN * 4);
    __bf16* outh = (__bf16*)walloc((size_t)MROWS * EMB * 2);   // [B,S,E]

    float* out1 = (float*)d_out;                               // [B,S,E]
    float* out2 = (float*)d_out + (size_t)MROWS * EMB;         // [B,Sq,Sk]

    // fp32 -> bf16 conversions
    int n4 = (MROWS * EMB) / 4;
    cvt_f32_to_bf16_kernel<<<(n4 + 255) / 256, 256, 0, stream>>>(
        (const float4*)query, (bf16x4*)qbf, n4);
    n4 = (EMB * EMB) / 4;
    cvt_f32_to_bf16_kernel<<<(n4 + 255) / 256, 256, 0, stream>>>((const float4*)Wq, (bf16x4*)wqb, n4);
    cvt_f32_to_bf16_kernel<<<(n4 + 255) / 256, 256, 0, stream>>>((const float4*)Wk, (bf16x4*)wkb, n4);
    cvt_f32_to_bf16_kernel<<<(n4 + 255) / 256, 256, 0, stream>>>((const float4*)Wv, (bf16x4*)wvb, n4);
    cvt_f32_to_bf16_kernel<<<(n4 + 255) / 256, 256, 0, stream>>>((const float4*)Wo, (bf16x4*)wob, n4);

    // projections (scale 1/sqrt(HD) folded into Q)
    dim3 ggrid(MROWS / 128, EMB / 128);
    gemm_bias_kernel<<<ggrid, 128, 0, stream>>>(qbf, wqb, bq, Qh, nullptr, 0.125f, 0);
    gemm_bias_kernel<<<ggrid, 128, 0, stream>>>(qbf, wkb, bk, Kh, nullptr, 1.0f, 0);
    gemm_bias_kernel<<<ggrid, 128, 0, stream>>>(qbf, wvb, bv, Vth, nullptr, 1.0f, 1);

    // online softmax stats
    softmax_stats_kernel<<<dim3(BATCH * HEADS, S_LEN / 512), 256, 0, stream>>>(Qh, Kh, Mst, Lst);

    // head-averaged attention probabilities
    attn_mean_kernel<<<dim3(S_LEN / 128, S_LEN / 128, BATCH), 128, 0, stream>>>(
        Qh, Kh, Mst, Lst, out2);

    // P @ V per head -> concatenated heads (bf16)
    attn_v_kernel<<<dim3(BATCH * HEADS, S_LEN / 256), 128, 0, stream>>>(
        Qh, Kh, Vth, Mst, Lst, outh);

    // output projection -> fp32
    gemm_bias_kernel<<<ggrid, 128, 0, stream>>>(outh, wob, bo, nullptr, out1, 1.0f, 2);
}
